// Attention_18923625906501
// MI455X (gfx1250) — matrix operand order repaired and checked by reading, not compiled, not run on hardware
//
#include <hip/hip_runtime.h>
#include <hip/hip_bf16.h>

// ---------------- problem constants ----------------
#define BATCH   4
#define CCH     256
#define HH      128
#define WWID    128
#define NHEADS  8
#define HDIM    32
#define KSZ     7
#define DILV    2
#define NBHV    3
#define HWSZ    (HH * WWID)

typedef __attribute__((ext_vector_type(16))) __bf16 v16bf;
typedef __attribute__((ext_vector_type(8)))  float  v8f;
typedef __attribute__((ext_vector_type(4)))  unsigned int u32x4;
typedef __attribute__((ext_vector_type(8)))  int          i32x8;
typedef __attribute__((ext_vector_type(4)))  int          i32x4;

// ---------------- NATTEN window helpers (match reference _window_start/_pb_start, d=2) ----
__device__ __forceinline__ int window_start(int i, int L) {
    int ni = i - NBHV * DILV;
    if (ni < 0) return i % DILV;
    if (i + NBHV * DILV >= L) {
        int imodd = i % DILV;
        int a = (L / DILV) * DILV;
        int bb = L - a;
        if (imodd < bb) return L - bb + imodd - 2 * NBHV * DILV;
        return a + imodd - KSZ * DILV;
    }
    return ni;
}
__device__ __forceinline__ int pb_start(int i, int L) {
    if (i - NBHV * DILV < 0) return KSZ - 1 - i / DILV;
    if (i + NBHV * DILV >= L) return (L - i - 1) / DILV;
    return NBHV;
}

// ---------------- WMMA GEMM: D[m,n] = sum_k A[m,k]*B[k,n] + bias[m] ----------------
// A: row-major [M,K] fp32 (weights, shared across batch; small, stays in L2/L0)
// B: row-major [K,N] fp32, batched with strideB (the big streaming operand)
// One wave -> one 16x16 tile via v_wmma_f32_16x16x32_bf16; 8 waves/WG -> 128 M-rows,
// all sharing one N-tile. The shared 32x16 fp32 B-slab is staged into LDS once per
// K-step by the Tensor Data Mover (tensor_load_to_lds + s_wait_tensorcnt) instead of
// being re-fetched by every wave (8x VMEM traffic reduction on the large operand).
// M % 128 == 0, N % 16 == 0, K % 32 == 0 (holds: M in {768,256}, N=16384, K=256).
__global__ __launch_bounds__(256) void gemm_bf16_wmma(
    const float* __restrict__ A, const float* __restrict__ B,
    const float* __restrict__ bias, float* __restrict__ D,
    int N, int Kd, size_t strideB, size_t strideD)
{
    __shared__ float tileB[32 * 16];   // one K-slab of B, shared by all 8 waves

    const int lane = threadIdx.x & 31;
    const int wave = threadIdx.x >> 5;
    const int hgrp = lane >> 4;     // half-wave group (0/1)
    const int l16  = lane & 15;

    const int m0   = blockIdx.y * 128 + wave * 16;   // wave tile row base
    const int rowA = m0 + l16;                        // A-fragment row for this lane
    const int n0   = blockIdx.x * 16;
    const int nIdx = n0 + l16;                        // B/D column for this lane

    const float* Bb = B + strideB * (size_t)blockIdx.z;
    float*       Db = D + strideD * (size_t)blockIdx.z;

    v8f acc = {};
    for (int kk = 0; kk < Kd; kk += 32) {
        __syncthreads();   // previous iteration's LDS reads complete before overwrite

#if __has_builtin(__builtin_amdgcn_tensor_load_to_lds)
        if (wave == 0) {
            // ---- Tensor DMA descriptor (D#), CDNA5 ISA ----
            // 2-D tile: tile_dim0=16 cols (contiguous), tile_dim1=32 rows,
            // row stride = N elements, data_size = 4 bytes.
            unsigned long long ga =
                (unsigned long long)(uintptr_t)(Bb + (size_t)kk * N + n0);
            unsigned lds_off = (unsigned)(uintptr_t)&tileB[0]; // ADDR[31:0] = LDS offset

            u32x4 g0;
            g0[0] = 1u;                                   // count=1, user descriptor
            g0[1] = lds_off;                              // lds_addr [63:32]
            g0[2] = (unsigned)(ga & 0xFFFFFFFFull);       // global_addr[31:0]
            g0[3] = (unsigned)((ga >> 32) & 0x1FFFFFFull) // global_addr[56:32]
                    | (2u << 30);                         // type=2 ("image")

            i32x8 g1;
            g1[0] = (int)(2u << 16);                      // data_size=2 -> 4 bytes
            g1[1] = (int)(((unsigned)N & 0xFFFFu) << 16); // tensor_dim0[15:0]
            g1[2] = (int)(((unsigned)N >> 16)             // tensor_dim0[31:16]
                    | (((unsigned)Kd & 0xFFFFu) << 16));  // tensor_dim1[15:0]
            g1[3] = (int)(((unsigned)Kd >> 16)            // tensor_dim1[31:16]
                    | (16u << 16));                       // tile_dim0 = 16
            g1[4] = 32;                                   // tile_dim1=32, tile_dim2=0
            g1[5] = (int)(unsigned)N;                     // tensor_dim0_stride[31:0]
            g1[6] = 0;                                    // stride0 hi | stride1 lo
            g1[7] = 0;

            i32x4 z4 = {};
#if defined(__clang_major__) && __clang_major__ >= 23
            i32x8 z8 = {};
            __builtin_amdgcn_tensor_load_to_lds(g0, g1, z4, z4, z8, 0);
#else
            __builtin_amdgcn_tensor_load_to_lds(g0, g1, z4, z4, 0);
#endif
            __builtin_amdgcn_s_wait_tensorcnt(0);
        }
#else
        // Fallback: cooperative load, 512 elements / 256 threads = 2 each
        {
            int t = threadIdx.x;
            tileB[t]       = Bb[(size_t)(kk + (t >> 4)) * N + n0 + (t & 15)];
            int t2 = t + 256;
            tileB[t2]      = Bb[(size_t)(kk + (t2 >> 4)) * N + n0 + (t2 & 15)];
        }
#endif
        __syncthreads();   // B slab visible to all 8 waves

        v16bf a, b;
        // A fragment: lane row rowA, K = kk + hgrp*8 + {0..7} and kk+16+hgrp*8+{0..7}
        const float* Arow = A + (size_t)rowA * Kd + kk + hgrp * 8;
        #pragma unroll
        for (int e = 0; e < 8; ++e) a[e]     = (__bf16)Arow[e];
        #pragma unroll
        for (int e = 0; e < 8; ++e) a[8 + e] = (__bf16)Arow[16 + e];

        // B fragment from LDS: lane col l16, K = hgrp*8 + {0..7} and 16+hgrp*8+{0..7}
        #pragma unroll
        for (int e = 0; e < 16; ++e)
            b[e] = (__bf16)tileB[(hgrp * 8 + (e & 7) + (e >> 3) * 16) * 16 + l16];

        if (kk + 32 < Kd)  // pull next K-slab toward L2 (global_prefetch_b8)
            __builtin_prefetch(&Bb[(size_t)(kk + 32 + hgrp * 16) * N + nIdx], 0, 3);

        acc = __builtin_amdgcn_wmma_f32_16x16x32_bf16(
                  false, a, false, b, (short)0, acc, false, false);
    }

    // C/D layout: VGPR r -> row m0 + hgrp*8 + r, col nIdx
    #pragma unroll
    for (int r = 0; r < 8; ++r) {
        const int gm = m0 + hgrp * 8 + r;
        Db[(size_t)gm * N + nIdx] = acc[r] + bias[gm];
    }
}

// ---------------- RoPE (in-place on q,k planes of qkv) ----------------
__global__ __launch_bounds__(256) void rope_kernel(
    float* __restrict__ qkv, const float* __restrict__ sinb, const float* __restrict__ cosb)
{
    size_t idx = (size_t)blockIdx.x * blockDim.x + threadIdx.x; // B*NH*(HD/2)*H*W
    int w  = (int)(idx % WWID);
    int h  = (int)((idx / WWID) % HH);
    int d2 = (int)((idx / HWSZ) % (HDIM / 2));
    int n  = (int)((idx / ((size_t)HWSZ * (HDIM / 2))) % NHEADS);
    int b  = (int)( idx / ((size_t)HWSZ * (HDIM / 2) * NHEADS));

    size_t sp = (size_t)h * WWID + w;
    size_t sc = sp * HDIM + 2 * d2;
    float c0 = cosb[sc], c1 = cosb[sc + 1];
    float s0 = sinb[sc], s1 = sinb[sc + 1];

    size_t qi = ((size_t)b * (3 * CCH) + n * HDIM + 2 * d2) * HWSZ + sp;
    size_t kv = qi + (size_t)CCH * HWSZ;
    float q0 = qkv[qi], q1 = qkv[qi + HWSZ];
    float k0 = qkv[kv], k1 = qkv[kv + HWSZ];

    const float scale = 0.17677669529663687f; // HD^-0.5
    // rotate_every_two: rot[2i] = -x[2i+1], rot[2i+1] = x[2i]
    qkv[qi]        = (q0 * c0 - q1 * s0) * scale;
    qkv[qi + HWSZ] = (q1 * c1 + q0 * s1) * scale;
    qkv[kv]        =  k0 * c0 - k1 * s0;
    qkv[kv + HWSZ] =  k1 * c1 + k0 * s1;
}

// ---------------- LePE: 5x5 depthwise conv (pad 2) on v planes ----------------
__global__ __launch_bounds__(256) void lepe_kernel(
    const float* __restrict__ qkv, const float* __restrict__ wl,
    const float* __restrict__ bl, float* __restrict__ lepe)
{
    size_t idx = (size_t)blockIdx.x * blockDim.x + threadIdx.x; // B*C*H*W
    int w = (int)(idx % WWID);
    int h = (int)((idx / WWID) % HH);
    int c = (int)((idx / HWSZ) % CCH);
    int b = (int)( idx / ((size_t)HWSZ * CCH));

    const float* vpl = qkv + ((size_t)b * (3 * CCH) + 2 * CCH + c) * HWSZ;
    float acc = bl[c];
    #pragma unroll
    for (int i = 0; i < 5; ++i) {
        int hy = h + i - 2;
        if (hy < 0 || hy >= HH) continue;
        #pragma unroll
        for (int j = 0; j < 5; ++j) {
            int wx = w + j - 2;
            if (wx < 0 || wx >= WWID) continue;
            acc += wl[c * 25 + i * 5 + j] * vpl[(size_t)hy * WWID + wx];
        }
    }
    lepe[idx] = acc;
}

// ---------------- dilated 7x7 neighborhood attention, online softmax ----------------
// One thread per (b, head, h, w); adds LePE and writes combined [B,C,H,W] fp32.
__global__ __launch_bounds__(256) void natten_kernel(
    const float* __restrict__ qkv, const float* __restrict__ rpb,
    const float* __restrict__ lepe, float* __restrict__ comb)
{
    size_t idx = (size_t)blockIdx.x * blockDim.x + threadIdx.x; // B*NH*H*W
    int w = (int)(idx % WWID);
    int h = (int)((idx / WWID) % HH);
    int n = (int)((idx / HWSZ) % NHEADS);
    int b = (int)( idx / ((size_t)HWSZ * NHEADS));

    const int sh = window_start(h, HH), sw = window_start(w, WWID);
    const int ph = pb_start(h, HH),     pw = pb_start(w, WWID);

    const size_t sp = (size_t)h * WWID + w;
    const float* qpl = qkv + ((size_t)b * (3 * CCH) + n * HDIM) * HWSZ;
    const float* kpl = qpl + (size_t)CCH * HWSZ;
    const float* vpl = kpl + (size_t)CCH * HWSZ;

    float q[HDIM];
    #pragma unroll
    for (int d = 0; d < HDIM; ++d) q[d] = qpl[sp + (size_t)d * HWSZ];

    float m = -1e30f, l = 0.f, o[HDIM];
    #pragma unroll
    for (int d = 0; d < HDIM; ++d) o[d] = 0.f;

    for (int ki = 0; ki < KSZ; ++ki) {
        const int hh = sh + ki * DILV;
        const float* rrow = rpb + ((size_t)n * (2 * KSZ - 1) + ph + ki) * (2 * KSZ - 1) + pw;
        for (int kj = 0; kj < KSZ; ++kj) {
            const int ww = sw + kj * DILV;
            const size_t nb = (size_t)hh * WWID + ww;
            float s = rrow[kj];
            #pragma unroll
            for (int d = 0; d < HDIM; ++d) s += q[d] * kpl[nb + (size_t)d * HWSZ];
            const float mn = fmaxf(m, s);
            const float p  = __expf(s - mn);
            const float rs = __expf(m - mn);
            l = l * rs + p;
            #pragma unroll
            for (int d = 0; d < HDIM; ++d)
                o[d] = o[d] * rs + p * vpl[nb + (size_t)d * HWSZ];
            m = mn;
        }
    }
    const float inv = 1.f / l;
    const size_t cb = ((size_t)b * CCH + n * HDIM) * HWSZ + sp;
    #pragma unroll
    for (int d = 0; d < HDIM; ++d)
        comb[cb + (size_t)d * HWSZ] = o[d] * inv + lepe[cb + (size_t)d * HWSZ];
}

// ---------------- host side ----------------
extern "C" void kernel_launch(void* const* d_in, const int* in_sizes, int n_in,
                              void* d_out, int out_size, void* d_ws, size_t ws_size,
                              hipStream_t stream) {
    const float* x      = (const float*)d_in[0];
    const float* sinb   = (const float*)d_in[1];
    const float* cosb   = (const float*)d_in[2];
    const float* w_qkv  = (const float*)d_in[3];
    const float* b_qkv  = (const float*)d_in[4];
    const float* w_lepe = (const float*)d_in[5];
    const float* b_lepe = (const float*)d_in[6];
    const float* w_proj = (const float*)d_in[7];
    const float* b_proj = (const float*)d_in[8];
    const float* rpb    = (const float*)d_in[9];
    float* out = (float*)d_out;

    // workspace: qkv fp32 | lepe fp32 | combined fp32
    float* qkv  = (float*)d_ws;
    float* lepe = qkv  + (size_t)BATCH * 3 * CCH * HWSZ;
    float* comb = lepe + (size_t)BATCH * CCH * HWSZ;

    // 1) QKV projection: M=768, N=16384 (per batch), K=256  -> WMMA bf16 + TDM
    dim3 g1(HWSZ / 16, (3 * CCH) / 128, BATCH);
    gemm_bf16_wmma<<<g1, 256, 0, stream>>>(
        w_qkv, x, b_qkv, qkv, HWSZ, CCH,
        (size_t)CCH * HWSZ, (size_t)3 * CCH * HWSZ);

    // 2) RoPE on q,k (in place)
    rope_kernel<<<(BATCH * NHEADS * (HDIM / 2) * HWSZ) / 256, 256, 0, stream>>>(
        qkv, sinb, cosb);

    // 3) LePE depthwise conv on v
    lepe_kernel<<<(BATCH * CCH * HWSZ) / 256, 256, 0, stream>>>(
        qkv, w_lepe, b_lepe, lepe);

    // 4) neighborhood attention + LePE add -> combined
    natten_kernel<<<(BATCH * NHEADS * HWSZ) / 256, 256, 0, stream>>>(
        qkv, rpb, lepe, comb);

    // 5) output projection: M=256, N=16384 (per batch), K=256 -> WMMA bf16 + TDM
    dim3 g2(HWSZ / 16, CCH / 128, BATCH);
    gemm_bf16_wmma<<<g2, 256, 0, stream>>>(
        w_proj, comb, b_proj, out, HWSZ, CCH,
        (size_t)CCH * HWSZ, (size_t)CCH * HWSZ);
}
